// HebbianBlock_39281770889407
// MI455X (gfx1250) — matrix operand order repaired and checked by reading, compiled, not run on hardware
//
#include <hip/hip_runtime.h>
#include <cstdint>

#define DEV_INLINE __device__ __forceinline__

typedef __attribute__((ext_vector_type(16))) __bf16 v16bf;
typedef __attribute__((ext_vector_type(8)))  __bf16 v8bf;
typedef __attribute__((ext_vector_type(8)))  float  v8f;
typedef __attribute__((ext_vector_type(4)))  unsigned int u32x4;
typedef __attribute__((ext_vector_type(8)))  int    i32x8;
typedef __attribute__((ext_vector_type(4)))  int    i32x4;

// ---------------- problem constants ----------------
constexpr int Bsz = 4, Tlen = 4096, DD = 1024, HDim = 64, Hn = 16, Cc = 64;
constexpr int BT = Bsz * Tlen;      // 16384 rows
constexpr int nChunk = Tlen / Cc;   // 64
constexpr int BH = Bsz * Hn;        // 64

// ---------------- table layout (floats) ----------------
constexpr int TBL_GAMMA = 0;
constexpr int TBL_GC    = 16;
constexpr int TBL_ALPHA = 32;
constexpr int TBL_GP    = 64;              // H*C
constexpr int TBL_GW    = 64 + 1024;       // H*C
constexpr int TBL_M     = 64 + 2048;       // H*C*C

// ---------------- workspace byte offsets ----------------
constexpr size_t OFF_OUT_BF = 0;                                        // BT*DD*2 (reused as READS_BF later)
constexpr size_t OFF_WW     = OFF_OUT_BF + (size_t)BT * DD * 2;         // DD*DD*2
constexpr size_t OFF_WR     = OFF_WW     + (size_t)DD * DD * 2;         // DD*DD*2
constexpr size_t OFF_VBF    = OFF_WR     + (size_t)DD * DD * 2;         // BT*DD*2
constexpr size_t OFF_READS  = OFF_VBF    + (size_t)BT * DD * 2;         // BT*DD*4
constexpr size_t OFF_U      = OFF_READS  + (size_t)BT * DD * 4;         // BH*nC*64*64*4
constexpr size_t OFF_TBL    = OFF_U      + (size_t)BH * nChunk * HDim * HDim * 4;

// ---------------- TDM availability ----------------
#if defined(__has_builtin)
#  if __has_builtin(__builtin_amdgcn_tensor_load_to_lds) && __has_builtin(__builtin_amdgcn_s_wait_tensorcnt)
#    define USE_TDM 1
#  else
#    define USE_TDM 0
#  endif
#else
#  define USE_TDM 0
#endif

// ---------------- small device helpers ----------------
DEV_INLINE v8f v8f_zero() {
  v8f z;
#pragma unroll
  for (int i = 0; i < 8; ++i) z[i] = 0.f;
  return z;
}

DEV_INLINE v8f wmma_bf16(v16bf a, v16bf b, v8f c) {
  // D = A(16x32 bf16) * B(32x16 bf16) + C(16x16 f32)
  return __builtin_amdgcn_wmma_f32_16x16x32_bf16(false, a, false, b, (short)0, c, false, false);
}

// A-fragment: lane holds K {k0..k0+15} with k0 = (lane>=16)*16
DEV_INLINE v16bf ld_afrag(const __bf16* rowp, int k0) {
  v8bf lo = *(const v8bf*)(rowp + k0);
  v8bf hi = *(const v8bf*)(rowp + k0 + 8);
  v16bf a;
#pragma unroll
  for (int i = 0; i < 8; ++i) { a[i] = lo[i]; a[i + 8] = hi[i]; }
  return a;
}

// convert 16 consecutive fp32 -> 16 bf16 (two v8bf stores)
DEV_INLINE void cvt16_store(__bf16* dst, const float* src) {
#pragma unroll
  for (int q = 0; q < 2; ++q) {
    float4 a = ((const float4*)src)[2 * q];
    float4 c = ((const float4*)src)[2 * q + 1];
    v8bf o;
    o[0] = (__bf16)a.x; o[1] = (__bf16)a.y; o[2] = (__bf16)a.z; o[3] = (__bf16)a.w;
    o[4] = (__bf16)c.x; o[5] = (__bf16)c.y; o[6] = (__bf16)c.z; o[7] = (__bf16)c.w;
    *(v8bf*)(dst + q * 8) = o;
  }
}

#if USE_TDM
// Issue one 2D TDM tile load: rows x cols bf16 elements, row stride (elements),
// packed contiguously (cols*2 bytes per row) at LDS byte offset lds_off.
// 6-arg toolchain variant: (u32x4 g0, i32x8 g1, i32x4 g2, i32x4 g3, i32x8 g4, i32 cpol)
DEV_INLINE void tdm_load_tile(const __bf16* gptr, unsigned lds_off,
                              int rows, int cols, int stride_elems) {
  unsigned long long ga = (unsigned long long)(uintptr_t)gptr;
  u32x4 g0;
  g0[0] = 1u;                                          // count=1, user mode
  g0[1] = lds_off;                                     // lds_addr
  g0[2] = (unsigned)(ga & 0xFFFFFFFFu);                // global_addr lo
  g0[3] = (unsigned)((ga >> 32) & 0x01FFFFFFu) | (2u << 30); // addr hi | type=2
  unsigned td0 = (unsigned)stride_elems;               // tensor_dim0 (elements)
  unsigned td1 = 0x00FFFFFFu;                          // tensor_dim1 (large, in-bounds tiles)
  unsigned long long s0 = (unsigned long long)stride_elems; // tensor_dim0_stride
  i32x8 g1;
  g1[0] = (int)(1u << 16);                             // data_size=1 (2 bytes)
  g1[1] = (int)((td0 & 0xFFFFu) << 16);                // barrier_addr=0 | td0[15:0]
  g1[2] = (int)((td0 >> 16) | ((td1 & 0xFFFFu) << 16));
  g1[3] = (int)((td1 >> 16) | ((unsigned)cols << 16)); // tile_dim0 = cols
  g1[4] = (int)((unsigned)rows);                       // tile_dim1 = rows, tile_dim2=0
  g1[5] = (int)(s0 & 0xFFFFFFFFu);
  g1[6] = (int)((s0 >> 32) & 0xFFFFu);                 // stride0 hi | stride1 lo16 = 0
  g1[7] = 0;
  i32x4 g2; g2[0] = 0; g2[1] = 0; g2[2] = 0; g2[3] = 0;
  i32x4 g3; g3[0] = 0; g3[1] = 0; g3[2] = 0; g3[3] = 0;
  i32x8 g4;
#pragma unroll
  for (int i = 0; i < 8; ++i) g4[i] = 0;
  __builtin_amdgcn_tensor_load_to_lds(g0, g1, g2, g3, g4, 0);
}
#endif

// ================= setup kernels =================
__global__ void k_tables(const float* __restrict__ decay,
                         const float* __restrict__ log_alpha,
                         float* __restrict__ tbl) {
  int h = blockIdx.x, p = threadIdx.x;
  float gamma = 1.f / (1.f + expf(-decay[h]));
  tbl[TBL_GP + h * Cc + p] = powf(gamma, (float)p);
  tbl[TBL_GW + h * Cc + p] = powf(gamma, (float)(Cc - 1 - p));
  if (p == 0) {
    tbl[TBL_GAMMA + h] = gamma;
    tbl[TBL_GC + h]    = powf(gamma, (float)Cc);
    tbl[TBL_ALPHA + h] = expf(log_alpha[h]);
  }
}

__global__ void k_mask(const float* __restrict__ decay, float* __restrict__ tbl) {
  int h = blockIdx.x, i = blockIdx.y, j = threadIdx.x;
  float gamma = 1.f / (1.f + expf(-decay[h]));
  float m = (i > j) ? powf(gamma, (float)(i - 1 - j)) : 0.f;
  tbl[TBL_M + (h * Cc + i) * Cc + j] = m;
}

__global__ void k_cvt_bf16(const float* __restrict__ s, __bf16* __restrict__ d, size_t n8) {
  size_t i = (size_t)blockIdx.x * blockDim.x + threadIdx.x;
  if (i >= n8) return;
  cvt16_store(d + i * 8, s + i * 8);
}

__global__ void k_cvt_reads(const float* __restrict__ s, __bf16* __restrict__ d,
                            const float* __restrict__ tbl) {
  size_t i = (size_t)blockIdx.x * blockDim.x + threadIdx.x;
  size_t e = i * 8;
  int col = (int)(e % DD);
  float alpha = tbl[TBL_ALPHA + (col >> 6)];
  float4 a = ((const float4*)(s + e))[0];
  float4 c = ((const float4*)(s + e))[1];
  v8bf o;
  o[0] = (__bf16)(a.x * alpha); o[1] = (__bf16)(a.y * alpha);
  o[2] = (__bf16)(a.z * alpha); o[3] = (__bf16)(a.w * alpha);
  o[4] = (__bf16)(c.x * alpha); o[5] = (__bf16)(c.y * alpha);
  o[6] = (__bf16)(c.z * alpha); o[7] = (__bf16)(c.w * alpha);
  *(v8bf*)(d + e) = o;
}

// ================= big GEMM: C[m,n] = sum_k A[m,k]*W[n,k] =================
DEV_INLINE void gemm_step(const __bf16* Ab, const __bf16* Bb, int lane,
                          int mbase, int nbase, v8f acc[4][2]) {
  v16bf af[4];
#pragma unroll
  for (int mt = 0; mt < 4; ++mt) {
    int row = mbase + mt * 16 + (lane & 15);
    af[mt] = ld_afrag(Ab + row * 32, (lane >> 4) * 16);
  }
#pragma unroll
  for (int nt = 0; nt < 2; ++nt) {
    int nr = nbase + nt * 16 + (lane & 15);
    v16bf bb = *(const v16bf*)(Bb + nr * 32 + (lane >> 4) * 16);
#pragma unroll
    for (int mt = 0; mt < 4; ++mt)
      acc[mt][nt] = wmma_bf16(af[mt], bb, acc[mt][nt]);
  }
}

template <int MODE>
__global__ __launch_bounds__(256)
void k_gemm_bf16(const __bf16* __restrict__ Ag, const __bf16* __restrict__ Bg,
                 const float* __restrict__ base, float* __restrict__ outF,
                 __bf16* __restrict__ outB) {
  constexpr int K = DD, N = DD;
  __shared__ alignas(16) __bf16 As[2][128 * 32];
  __shared__ alignas(16) __bf16 Bs[2][128 * 32];

  const int tid = threadIdx.x, lane = tid & 31, wave = tid >> 5;
  const int mblk = blockIdx.x, nblk = blockIdx.y;
  const int wm = wave & 1, wn = wave >> 1;         // 2 x 4 wave grid
  const int mbase = wm * 64, nbase = wn * 32;

  v8f acc[4][2];
#pragma unroll
  for (int mt = 0; mt < 4; ++mt)
#pragma unroll
    for (int nt = 0; nt < 2; ++nt) acc[mt][nt] = v8f_zero();

#if USE_TDM
  const unsigned aoff0 = (unsigned)(uintptr_t)&As[0][0];
  const unsigned aoff1 = (unsigned)(uintptr_t)&As[1][0];
  const unsigned boff0 = (unsigned)(uintptr_t)&Bs[0][0];
  const unsigned boff1 = (unsigned)(uintptr_t)&Bs[1][0];
  const __bf16* Abase = Ag + (size_t)(mblk * 128) * K;
  const __bf16* Bbase = Bg + (size_t)(nblk * 128) * K;
  if (wave == 0) {
    tdm_load_tile(Abase + 0,  aoff0, 128, 32, K);
    tdm_load_tile(Bbase + 0,  boff0, 128, 32, K);
    tdm_load_tile(Abase + 32, aoff1, 128, 32, K);
    tdm_load_tile(Bbase + 32, boff1, 128, 32, K);
  }
  for (int ks = 0; ks < K / 32; ++ks) {
    if (wave == 0) __builtin_amdgcn_s_wait_tensorcnt(2);  // oldest pair done (in-order)
    __syncthreads();
    gemm_step(&As[ks & 1][0], &Bs[ks & 1][0], lane, mbase, nbase, acc);
    __syncthreads();
    if (wave == 0 && ks + 2 < K / 32) {
      tdm_load_tile(Abase + (ks + 2) * 32, (ks & 1) ? aoff1 : aoff0, 128, 32, K);
      tdm_load_tile(Bbase + (ks + 2) * 32, (ks & 1) ? boff1 : boff0, 128, 32, K);
    }
  }
#else
  for (int ks = 0; ks < K / 32; ++ks) {
    __syncthreads();
    {
      int r = tid >> 1, c0 = (tid & 1) * 16;
      const __bf16* ga = Ag + (size_t)(mblk * 128 + r) * K + ks * 32 + c0;
      *(v8bf*)&As[0][r * 32 + c0]     = *(const v8bf*)ga;
      *(v8bf*)&As[0][r * 32 + c0 + 8] = *(const v8bf*)(ga + 8);
      const __bf16* gb = Bg + (size_t)(nblk * 128 + r) * K + ks * 32 + c0;
      *(v8bf*)&Bs[0][r * 32 + c0]     = *(const v8bf*)gb;
      *(v8bf*)&Bs[0][r * 32 + c0 + 8] = *(const v8bf*)(gb + 8);
    }
    __syncthreads();
    gemm_step(&As[0][0], &Bs[0][0], lane, mbase, nbase, acc);
  }
#endif

#pragma unroll
  for (int mt = 0; mt < 4; ++mt)
#pragma unroll
    for (int nt = 0; nt < 2; ++nt) {
      int gcol = nblk * 128 + nbase + nt * 16 + (lane & 15);
#pragma unroll
      for (int r = 0; r < 8; ++r) {
        int grow = mblk * 128 + mbase + mt * 16 + r + 8 * (lane >> 4);
        size_t idx = (size_t)grow * N + gcol;
        if (MODE == 0) outB[idx] = (__bf16)acc[mt][nt][r];
        else           outF[idx] = base[idx] + acc[mt][nt][r];
      }
    }
}

// ================= chunk-local kernel =================
// grid (BH, nChunk); per WG: S = rk wk^T, intra = (S.M) v, U = (v*g_w)^T wk
__global__ __launch_bounds__(256)
void k_chunk(const float* __restrict__ outp, const __bf16* __restrict__ vbf,
             float* __restrict__ readsF, float* __restrict__ Ubuf,
             const float* __restrict__ tbl) {
  const int bh = blockIdx.x, chunk = blockIdx.y;
  const int b = bh >> 4, h = bh & 15;
  __shared__ alignas(16) __bf16 RK[64 * 64];   // rk_c  [c][k]
  __shared__ alignas(16) __bf16 WK[64 * 64];   // wk_c  [c][k]
  __shared__ alignas(16) __bf16 WKT[64 * 64];  // wk_c^T [j][c]
  __shared__ alignas(16) __bf16 VT[64 * 64];   // v_c^T  [dd][c]
  __shared__ alignas(16) __bf16 VTW[64 * 64];  // (v_c*g_w)^T [i][c]
  __shared__ alignas(16) __bf16 SP[64 * 64];   // masked S, bf16
  __shared__ alignas(16) float  MS[64 * 64];   // decay mask tile (fp32, coalesced staged)

  const int tid = threadIdx.x, lane = tid & 31, wave = tid >> 5;
  const int mt = wave >> 1, ntp = (wave & 1) * 2;

  {  // staging: 4 threads per row, 16 cols each
    int row = tid >> 2, c0 = (tid & 3) * 16;
    size_t t = (size_t)b * Tlen + (size_t)chunk * Cc + row;
    const float* srk = outp + t * DD + h * HDim + c0;
    cvt16_store(&RK[row * 64 + c0], srk);
    float tmpw[16];
    if (chunk == 0 && row == 0) {
#pragma unroll
      for (int i = 0; i < 16; ++i) tmpw[i] = 0.f;
    } else {
#pragma unroll
      for (int i = 0; i < 16; ++i) tmpw[i] = srk[i - DD];
    }
    cvt16_store(&WK[row * 64 + c0], tmpw);
#pragma unroll
    for (int i = 0; i < 16; ++i) WKT[(c0 + i) * 64 + row] = (__bf16)tmpw[i];
    const __bf16* sv = vbf + t * DD + h * HDim + c0;
    v8bf v0 = ((const v8bf*)sv)[0];
    v8bf v1 = ((const v8bf*)sv)[1];
    float gwc = tbl[TBL_GW + h * Cc + row];
#pragma unroll
    for (int i = 0; i < 8; ++i) {
      VT[(c0 + i) * 64 + row] = v0[i];
      VT[(c0 + 8 + i) * 64 + row] = v1[i];
      VTW[(c0 + i) * 64 + row] = (__bf16)((float)v0[i] * gwc);
      VTW[(c0 + 8 + i) * 64 + row] = (__bf16)((float)v1[i] * gwc);
    }
    // coalesced stage of the head's 64x64 decay mask into LDS
    const float* msrc = tbl + TBL_M + (size_t)h * Cc * Cc + row * 64 + c0;
#pragma unroll
    for (int q = 0; q < 4; ++q)
      ((float4*)&MS[row * 64 + c0])[q] = ((const float4*)msrc)[q];
  }
  __syncthreads();

  // ---- S = rk wk^T, apply decay mask, store bf16 ----
  v8f accS[2] = { v8f_zero(), v8f_zero() };
#pragma unroll
  for (int ks = 0; ks < 2; ++ks) {
    int arow = mt * 16 + (lane & 15);
    v16bf a = ld_afrag(&RK[arow * 64], ks * 32 + (lane >> 4) * 16);
    int kb = ks * 32 + (lane >> 4) * 16;
#pragma unroll
    for (int tt = 0; tt < 2; ++tt) {
      int nr = (ntp + tt) * 16 + (lane & 15);
      v16bf bb = *(const v16bf*)&WK[nr * 64 + kb];
      accS[tt] = wmma_bf16(a, bb, accS[tt]);
    }
  }
#pragma unroll
  for (int tt = 0; tt < 2; ++tt) {
    int j = (ntp + tt) * 16 + (lane & 15);
#pragma unroll
    for (int r = 0; r < 8; ++r) {
      int i = mt * 16 + r + 8 * (lane >> 4);
      float m = MS[i * 64 + j];
      SP[i * 64 + j] = (__bf16)(accS[tt][r] * m);
    }
  }
  __syncthreads();

  // ---- intra = SP @ v_c  and  U = (v*g_w)^T @ wk ----
  v8f accI[2] = { v8f_zero(), v8f_zero() };
  v8f accU[2] = { v8f_zero(), v8f_zero() };
#pragma unroll
  for (int ks = 0; ks < 2; ++ks) {
    int arow = mt * 16 + (lane & 15);
    int k0 = ks * 32 + (lane >> 4) * 16;
    v16bf aS = ld_afrag(&SP[arow * 64], k0);
    v16bf aW = ld_afrag(&VTW[arow * 64], k0);
    int kb = ks * 32 + (lane >> 4) * 16;
#pragma unroll
    for (int tt = 0; tt < 2; ++tt) {
      int nr = (ntp + tt) * 16 + (lane & 15);
      v16bf bV = *(const v16bf*)&VT[nr * 64 + kb];
      v16bf bK = *(const v16bf*)&WKT[nr * 64 + kb];
      accI[tt] = wmma_bf16(aS, bV, accI[tt]);
      accU[tt] = wmma_bf16(aW, bK, accU[tt]);
    }
  }
#pragma unroll
  for (int tt = 0; tt < 2; ++tt) {
    int n = (ntp + tt) * 16 + (lane & 15);
#pragma unroll
    for (int r = 0; r < 8; ++r) {
      int m = mt * 16 + r + 8 * (lane >> 4);
      size_t t = (size_t)b * Tlen + (size_t)chunk * Cc + m;
      readsF[t * DD + h * HDim + n] = accI[tt][r];
      Ubuf[((size_t)(bh * nChunk + chunk)) * 4096 + m * 64 + n] = accU[tt][r];
    }
  }
}

// ================= sequential state kernel =================
__global__ __launch_bounds__(256)
void k_scan(const float* __restrict__ outp, const float* __restrict__ Ubuf,
            float* __restrict__ readsF, const float* __restrict__ tbl) {
  const int bh = blockIdx.x;
  const int b = bh >> 4, h = bh & 15;
  const float gc = tbl[TBL_GC + h];
  const float* gp = tbl + TBL_GP + h * Cc;
  __shared__ float Wf[64 * 64];
  __shared__ alignas(16) __bf16 Wb[64 * 64];   // W as N-major operand: [i][j]
  __shared__ alignas(16) __bf16 RK[64 * 64];
  const int tid = threadIdx.x, lane = tid & 31, wave = tid >> 5;
  const int mt = wave >> 1, ntp = (wave & 1) * 2;

  for (int e = tid; e < 4096; e += 256) Wf[e] = 0.f;
  __syncthreads();

  for (int chunk = 0; chunk < nChunk; ++chunk) {
    if (chunk > 0) {
      {
        int row = tid >> 2, c0 = (tid & 3) * 16;
        const float* s = outp + ((size_t)b * Tlen + (size_t)chunk * Cc + row) * DD + h * HDim + c0;
        cvt16_store(&RK[row * 64 + c0], s);
      }
      for (int e = tid; e < 4096; e += 256) Wb[e] = (__bf16)Wf[e];
      __syncthreads();

      v8f acc[2] = { v8f_zero(), v8f_zero() };
#pragma unroll
      for (int ks = 0; ks < 2; ++ks) {
        int arow = mt * 16 + (lane & 15);
        v16bf a = ld_afrag(&RK[arow * 64], ks * 32 + (lane >> 4) * 16);
        int kb = ks * 32 + (lane >> 4) * 16;
#pragma unroll
        for (int tt = 0; tt < 2; ++tt) {
          int nr = (ntp + tt) * 16 + (lane & 15);
          v16bf bb = *(const v16bf*)&Wb[nr * 64 + kb];
          acc[tt] = wmma_bf16(a, bb, acc[tt]);
        }
      }
#pragma unroll
      for (int tt = 0; tt < 2; ++tt) {
        int n = (ntp + tt) * 16 + (lane & 15);
#pragma unroll
        for (int r = 0; r < 8; ++r) {
          int c = mt * 16 + r + 8 * (lane >> 4);
          size_t idx = ((size_t)b * Tlen + (size_t)chunk * Cc + c) * DD + h * HDim + n;
          readsF[idx] += acc[tt][r] * gp[c];
        }
      }
      __syncthreads();
    }
    const float* Uc = Ubuf + ((size_t)(bh * nChunk + chunk)) * 4096;
    for (int e = tid; e < 4096; e += 256) Wf[e] = gc * Wf[e] + Uc[e];
    __syncthreads();
  }
}

// ================= launch =================
extern "C" void kernel_launch(void* const* d_in, const int* in_sizes, int n_in,
                              void* d_out, int out_size, void* d_ws, size_t ws_size,
                              hipStream_t stream) {
  const float* outp  = (const float*)d_in[0];
  const float* Ww    = (const float*)d_in[1];
  const float* Wr    = (const float*)d_in[2];
  const float* decay = (const float*)d_in[3];
  const float* la    = (const float*)d_in[4];
  float* y = (float*)d_out;
  char* ws = (char*)d_ws;

  __bf16* OUT_BF   = (__bf16*)(ws + OFF_OUT_BF);
  __bf16* READS_BF = (__bf16*)(ws + OFF_OUT_BF);  // alias: OUT_BF free after v-GEMM
  __bf16* WW_BF    = (__bf16*)(ws + OFF_WW);
  __bf16* WR_BF    = (__bf16*)(ws + OFF_WR);
  __bf16* V_BF     = (__bf16*)(ws + OFF_VBF);
  float*  READS_F  = (float*)(ws + OFF_READS);
  float*  U_F      = (float*)(ws + OFF_U);
  float*  TBL      = (float*)(ws + OFF_TBL);

  k_tables<<<dim3(Hn), dim3(Cc), 0, stream>>>(decay, la, TBL);
  k_mask<<<dim3(Hn, Cc), dim3(Cc), 0, stream>>>(decay, TBL);

  k_cvt_bf16<<<dim3((unsigned)((size_t)BT * DD / 8 / 256)), dim3(256), 0, stream>>>(
      outp, OUT_BF, (size_t)BT * DD / 8);
  k_cvt_bf16<<<dim3((unsigned)((size_t)DD * DD / 8 / 256)), dim3(256), 0, stream>>>(
      Ww, WW_BF, (size_t)DD * DD / 8);
  k_cvt_bf16<<<dim3((unsigned)((size_t)DD * DD / 8 / 256)), dim3(256), 0, stream>>>(
      Wr, WR_BF, (size_t)DD * DD / 8);

  // v = out @ W_write^T  (bf16 result)
  k_gemm_bf16<0><<<dim3(BT / 128, DD / 128), dim3(256), 0, stream>>>(
      OUT_BF, WW_BF, nullptr, nullptr, V_BF);

  // chunk-local: intra + U
  k_chunk<<<dim3(BH, nChunk), dim3(256), 0, stream>>>(outp, V_BF, READS_F, U_F, TBL);

  // sequential inter-chunk state pass
  k_scan<<<dim3(BH), dim3(256), 0, stream>>>(outp, U_F, READS_F, TBL);

  // reads * alpha -> bf16
  k_cvt_reads<<<dim3((unsigned)((size_t)BT * DD / 8 / 256)), dim3(256), 0, stream>>>(
      READS_F, READS_BF, TBL);

  // y = out + reads @ W_read^T  (fp32 result)
  k_gemm_bf16<1><<<dim3(BT / 128, DD / 128), dim3(256), 0, stream>>>(
      READS_BF, WR_BF, outp, y, nullptr);
}
